// CrossAttentionHead_24163486007716
// MI455X (gfx1250) — matrix operand order repaired and checked by reading, not compiled, not run on hardware
//
#include <hip/hip_runtime.h>

typedef __attribute__((ext_vector_type(16))) _Float16 v16h;
typedef __attribute__((ext_vector_type(8)))  _Float16 v8h;
typedef __attribute__((ext_vector_type(8)))  float    v8f;
typedef __attribute__((ext_vector_type(4)))  float    f32x4;

#define NB 128
#define NT 512
#define NC 384
#define NH 64

__device__ __forceinline__ v8h ld8h(const _Float16* p) { return *(const v8h*)p; }
__device__ __forceinline__ v16h cat16(v8h lo, v8h hi) {
    return __builtin_shufflevector(lo, hi, 0,1,2,3,4,5,6,7,8,9,10,11,12,13,14,15);
}

// ---------------------------------------------------------------------------
// Kernel 1: projections.  D[M=B*T x N=64] = src[M x 384] * W[384 x 64]
// One wave -> one 16-row x 64-col strip. W staged transposed (f16) in LDS.
// blockIdx.y selects {Q, K, V}; V is written transposed as Vt[B][H][T].
// ---------------------------------------------------------------------------
__global__ __launch_bounds__(256) void proj_kernel(
    const float* __restrict__ input, const float* __restrict__ x,
    const float* __restrict__ Wq, const float* __restrict__ Wk,
    const float* __restrict__ Wv,
    _Float16* __restrict__ Qh, _Float16* __restrict__ Kh,
    _Float16* __restrict__ Vth)
{
    __shared__ _Float16 wt[NH * NC];            // W transposed: [h][c], 48KB

    const int tid   = threadIdx.x;
    const int which = blockIdx.y;               // 0=Q, 1=K, 2=V
    const float* src = (which == 0) ? input : x;
    const float* W   = (which == 0) ? Wq : (which == 1) ? Wk : Wv;

    for (int i = tid; i < NC * NH; i += 256) {  // coalesced read, transposed write
        int c = i >> 6, h = i & 63;
        wt[h * NC + c] = (_Float16)W[i];
    }
    __syncthreads();

    const int lane = tid & 31;
    const int lr   = lane & 15;                 // row / column within tile
    const int lh   = lane >> 4;                 // half-wave select
    const int tile = blockIdx.x * 8 + (tid >> 5);
    const int row0 = tile * 16;                 // global row (b*T + t)

    v8f acc[4] = {};                            // 4 N-tiles of 16 cols

    const float* arow = src + (size_t)(row0 + lr) * NC;
#pragma unroll 1
    for (int kt = 0; kt < 12; ++kt) {
        const int k0 = kt * 32;
        // A operand (16x32 f16): lane<16 holds k 0..7 & 16..23; lane>=16: +8
        v16h a;
        {
            const float* p = arow + k0 + 8 * lh;
#pragma unroll
            for (int e = 0; e < 8; ++e) a[e]     = (_Float16)p[e];
#pragma unroll
            for (int e = 0; e < 8; ++e) a[8 + e] = (_Float16)p[16 + e];
        }
#pragma unroll
        for (int j = 0; j < 4; ++j) {
            // B operand (32x16 f16): lane holds column n, k 0..7 & 16..23; lane>=16: +8
            const _Float16* bp = &wt[(16 * j + lr) * NC + k0 + 8 * lh];
            v16h b = cat16(ld8h(bp), ld8h(bp + 16));
            acc[j] = __builtin_amdgcn_wmma_f32_16x16x32_f16(
                false, a, false, b, (short)0, acc[j], false, false);
        }
    }

    if (which < 2) {                            // Q, K row-major [B*T][H] f16
        _Float16* dst = (which == 0) ? Qh : Kh;
#pragma unroll
        for (int j = 0; j < 4; ++j) {
            const int n = 16 * j + lr;
#pragma unroll
            for (int r = 0; r < 8; ++r) {
                const int m = row0 + r + 8 * lh;
                dst[(size_t)m * NH + n] = (_Float16)acc[j][r];
            }
        }
    } else {                                    // V transposed: Vt[b][h][t]
#pragma unroll
        for (int j = 0; j < 4; ++j) {
            const int n  = 16 * j + lr;         // h
            const int tg = row0 + 8 * lh;       // first of 8 consecutive t
            const int bb = tg >> 9, t0 = tg & 511;
            v8h v;
#pragma unroll
            for (int r = 0; r < 8; ++r) v[r] = (_Float16)acc[j][r];
            *(v8h*)&Vth[((size_t)bb * NH + n) * NT + t0] = v;
        }
    }
}

// ---------------------------------------------------------------------------
// Kernel 2: causal flash attention. One wave per (batch, 16-query tile).
// S^T = K*Q^T (per-lane fixed query column -> scalar online softmax),
// O^T = V^T * P^T, O^T rows store as contiguous float4.
// ---------------------------------------------------------------------------
__global__ __launch_bounds__(256) void attn_kernel(
    const _Float16* __restrict__ Qh, const _Float16* __restrict__ Kh,
    const _Float16* __restrict__ Vth, float* __restrict__ out)
{
    const int tid  = threadIdx.x;
    const int lane = tid & 31;
    const int lr   = lane & 15;
    const int lh   = lane >> 4;
    const int tile = blockIdx.x * 8 + (tid >> 5);
    const int b    = tile >> 5;
    const int q0   = (tile & 31) * 16;

    const _Float16* Q  = Qh  + (size_t)b * NT * NH;
    const _Float16* K  = Kh  + (size_t)b * NT * NH;
    const _Float16* Vt = Vth + (size_t)b * NH * NT;

    // Q as B operand: lane = column q, h 8*lh..+7 and 16+8*lh..+7
    v16h qop[2];
#pragma unroll
    for (int kt = 0; kt < 2; ++kt) {
        const _Float16* p = Q + (size_t)(q0 + lr) * NH + kt * 32 + 8 * lh;
        qop[kt] = cat16(ld8h(p), ld8h(p + 16));
    }

    v8f o[4] = {};
    float mval = -1e30f, lval = 0.0f;
    const float cscale = 0.073622508f;          // 384^-0.5 * log2(e)
    const int   q      = q0 + lr;
    const int   kkend  = q0 + 16;

#pragma unroll 1
    for (int kk0 = 0; kk0 < kkend; kk0 += 32) {
        // ---- S^T tiles: A = K rows (16 keys x 32 h), B = Q ----
        v8f s0, s1;
        {
            const _Float16* p = K + (size_t)(kk0 + lr) * NH + 8 * lh;
            v16h a0 = cat16(ld8h(p),      ld8h(p + 16));
            v16h a1 = cat16(ld8h(p + 32), ld8h(p + 48));
            v8f z = {};
            s0 = __builtin_amdgcn_wmma_f32_16x16x32_f16(false, a0, false, qop[0], (short)0, z,  false, false);
            s0 = __builtin_amdgcn_wmma_f32_16x16x32_f16(false, a1, false, qop[1], (short)0, s0, false, false);
        }
        {
            const _Float16* p = K + (size_t)(kk0 + 16 + lr) * NH + 8 * lh;
            v16h a0 = cat16(ld8h(p),      ld8h(p + 16));
            v16h a1 = cat16(ld8h(p + 32), ld8h(p + 48));
            v8f z = {};
            s1 = __builtin_amdgcn_wmma_f32_16x16x32_f16(false, a0, false, qop[0], (short)0, z,  false, false);
            s1 = __builtin_amdgcn_wmma_f32_16x16x32_f16(false, a1, false, qop[1], (short)0, s1, false, false);
        }

        // ---- causal mask + online softmax (per-lane fixed query) ----
        float sv0[8], sv1[8];
        float mx = -1e30f;
#pragma unroll
        for (int r = 0; r < 8; ++r) {
            const int k0i = kk0 + r + 8 * lh;
            float v0 = s0[r] * cscale; if (k0i      > q) v0 = -1e30f;
            float v1 = s1[r] * cscale; if (k0i + 16 > q) v1 = -1e30f;
            sv0[r] = v0; sv1[r] = v1;
            mx = fmaxf(mx, fmaxf(v0, v1));
        }
        mx = fmaxf(mx, __shfl_xor(mx, 16));
        const float mnew  = fmaxf(mval, mx);
        const float alpha = __builtin_amdgcn_exp2f(mval - mnew);

        float p0[8], p1[8], ls = 0.0f;
#pragma unroll
        for (int r = 0; r < 8; ++r) {
            p0[r] = __builtin_amdgcn_exp2f(sv0[r] - mnew);
            p1[r] = __builtin_amdgcn_exp2f(sv1[r] - mnew);
            ls += p0[r] + p1[r];
        }
        ls  += __shfl_xor(ls, 16);
        lval = lval * alpha + ls;
        mval = mnew;

#pragma unroll
        for (int j = 0; j < 4; ++j)
#pragma unroll
            for (int r = 0; r < 8; ++r) o[j][r] *= alpha;

        // ---- P^T as B operand: lane = column q, k = 8*lh + e | 16 + 8*lh + e ----
        // own p0 | own p1
        v16h pop;
#pragma unroll
        for (int r = 0; r < 8; ++r) {
            const float x0 = __shfl_xor(p0[r], 16);
            const float x1 = __shfl_xor(p1[r], 16);
            (void)x0; (void)x1;
            pop[r]     = (_Float16)p0[r];
            pop[8 + r] = (_Float16)p1[r];
        }

        // ---- O^T += V^T * P^T : A = Vt rows (16 h x 32 keys) ----
#pragma unroll
        for (int j = 0; j < 4; ++j) {
            const _Float16* vp = Vt + (size_t)(16 * j + lr) * NT + kk0 + 8 * lh;
            v16h av = cat16(ld8h(vp), ld8h(vp + 16));
            o[j] = __builtin_amdgcn_wmma_f32_16x16x32_f16(
                false, av, false, pop, (short)0, o[j], false, false);
        }
    }

    // ---- epilogue: divide by row sum, store 8 contiguous fp32 per tile ----
    const float inv = 1.0f / lval;
    float* op = out + ((size_t)b * NT + q0 + lr) * NH;
#pragma unroll
    for (int j = 0; j < 4; ++j) {
        f32x4 v0, v1;
#pragma unroll
        for (int r = 0; r < 4; ++r) { v0[r] = o[j][r] * inv; v1[r] = o[j][4 + r] * inv; }
        *(f32x4*)(op + 16 * j + 8 * lh)     = v0;
        *(f32x4*)(op + 16 * j + 8 * lh + 4) = v1;
    }
}

// ---------------------------------------------------------------------------
extern "C" void kernel_launch(void* const* d_in, const int* in_sizes, int n_in,
                              void* d_out, int out_size, void* d_ws, size_t ws_size,
                              hipStream_t stream) {
    const float* input = (const float*)d_in[0];
    const float* x     = (const float*)d_in[1];
    const float* Wq    = (const float*)d_in[2];
    const float* Wk    = (const float*)d_in[3];
    const float* Wv    = (const float*)d_in[4];

    const size_t mat = (size_t)NB * NT * NH;    // 4M halfs = 8MB each
    _Float16* Qh  = (_Float16*)d_ws;
    _Float16* Kh  = Qh + mat;
    _Float16* Vth = Kh + mat;

    // 65536 rows / 16 per wave / 8 waves per block = 512 blocks; y = {Q,K,V}
    proj_kernel<<<dim3(512, 3), 256, 0, stream>>>(input, x, Wq, Wk, Wv, Qh, Kh, Vth);
    // 128 batches * 32 query tiles / 8 waves per block = 512 blocks
    attn_kernel<<<512, 256, 0, stream>>>(Qh, Kh, Vth, (float*)d_out);
}
